// NpiLoss_84842783965555
// MI455X (gfx1250) — matrix operand order repaired and checked by reading, not compiled, not run on hardware
//
#include <hip/hip_runtime.h>
#include <hip/hip_bf16.h>

typedef __attribute__((ext_vector_type(16))) _Float16 v16h;
typedef __attribute__((ext_vector_type(8)))  _Float16 v8h;
typedef __attribute__((ext_vector_type(8)))  float    v8f;

#define N2 4096
#define M  64
#define BLK 64   // rows per scan block (4x4 tiles of 16x16 across 16 waves)

// ---------------------------------------------------------------------------
// Stage 1: blocked linear recurrence (single persistent workgroup, 16 waves).
// pred[i,m] = rt[i,m] * sum_{j<=i-2} si[i-j] * pred[j,m]
// Cross-block history = Toeplitz matmul via v_wmma_f32_16x16x32_f16.
// A operand comes from a reversed+pre-masked si table so every lane does
// branch-free contiguous loads; B operand is 16B-aligned b128 loads from the
// column-major f16 pred mirror. In-block tail: 64 threads, one column each.
// ---------------------------------------------------------------------------
__global__ __launch_bounds__(512) void stage1_kernel(
    const float* __restrict__ rt, const float* __restrict__ si,
    _Float16* __restrict__ sirev, float* __restrict__ predT_f,
    _Float16* __restrict__ predT_h)
{
    __shared__ float Base[BLK][BLK + 1];     // cross-block contributions
    __shared__ float PredBlk[BLK][BLK + 1];  // current block pred values
    __shared__ float sib[BLK];               // si[0..63] (in-block taps)

    const int tid = threadIdx.x;

    // Reversed, pre-masked f16 si table: sirev[t] = (4095-t>=2) ? si[4095-t] : 0
    // Reversal turns the Toeplitz A accesses into ascending contiguous runs.
    for (int t = tid; t < N2; t += 512) {
        int v = (N2 - 1) - t;
        sirev[t] = (v >= 2) ? (_Float16)si[v] : (_Float16)0.0f;
    }
    if (tid < BLK) sib[tid] = si[tid];
    __threadfence();
    __syncthreads();

    const int lane    = tid & 31;
    const int wave    = tid >> 5;             // 0..15
    const int rowbase = (wave >> 2) << 4;     // tile row: 0,16,32,48
    const int colbase = (wave & 3) << 4;      // tile col: 0,16,32,48
    const int row     = lane & 15;
    const int hi      = lane >> 4;            // lane half selects K sub-range

    for (int b = 0; b < N2; b += BLK) {
        // ---- Phase A: Base[i-b][m] = sum_{j<b, j<=i-2} si[i-j]*pred[j,m] ----
        v8f acc = {};
        const int gi = b + rowbase + row;     // global row index i
        // A: a[e] = sirev[(4095-gi) + k0 + K(e)], K(e) = e + hi*16
        const _Float16* arow = sirev + (N2 - 1 - gi) + hi * 16;
        // B: bv[e] = predT_h[n*N2 + k0 + hi*16 + e], contiguous 16 elements
        const _Float16* brow = predT_h + (colbase + row) * N2 + hi * 16;

        for (int k0 = 0; k0 < b; k0 += 2 * 32) {   // b is a multiple of 64
            #pragma unroll
            for (int kk = 0; kk < 64; kk += 32) {
                v8h alo, ahi;                       // 2B-aligned -> memcpy
                __builtin_memcpy(&alo, arow + k0 + kk, 16);
                __builtin_memcpy(&ahi, arow + k0 + kk + 8, 16);
                v8h blo = *(const v8h*)(brow + k0 + kk);       // 16B aligned
                v8h bhi = *(const v8h*)(brow + k0 + kk + 8);   // 16B aligned
                v16h a  = __builtin_shufflevector(alo, ahi,
                          0,1,2,3,4,5,6,7,8,9,10,11,12,13,14,15);
                v16h bv = __builtin_shufflevector(blo, bhi,
                          0,1,2,3,4,5,6,7,8,9,10,11,12,13,14,15);
                acc = __builtin_amdgcn_wmma_f32_16x16x32_f16(
                    false, a, false, bv, (short)0, acc, false, false);
            }
        }
        // C/D layout: VGPR r -> M = r + hi*8, N = lane&15
        #pragma unroll
        for (int r = 0; r < 8; ++r)
            Base[rowbase + hi * 8 + r][colbase + row] = acc[r];
        __syncthreads();

        // ---- Phase B: sequential in-block tail, one thread per column ----
        if (tid < M) {
            const int m = tid;
            for (int ii = 0; ii < BLK; ++ii) {
                float S = Base[ii][m];
                for (int jj = 0; jj + 2 <= ii; ++jj)
                    S = fmaf(sib[ii - jj], PredBlk[jj][m], S);
                float val = rt[(b + ii) * M + m] * S;
                PredBlk[ii][m] = val;
                predT_f[m * N2 + b + ii] = val;
                predT_h[m * N2 + b + ii] = (_Float16)val;
            }
            __threadfence();   // make predT_h visible to next block's WMMA loads
        }
        __syncthreads();
    }
}

// ---------------------------------------------------------------------------
// Gather convolution kernels: fmT[m][k] = (k<2) ? 0 : f[k, idx[m]]
// ---------------------------------------------------------------------------
__global__ __launch_bounds__(256) void fmT_kernel(
    const float* __restrict__ f, const int* __restrict__ idx,
    float* __restrict__ fmT)
{
    int t = blockIdx.x * 256 + threadIdx.x;   // 0 .. 64*4096-1
    int m = t >> 12;
    int k = t & (N2 - 1);
    fmT[m * N2 + k] = (k < 2) ? 0.0f : f[k * M + idx[m]];
}

// ---------------------------------------------------------------------------
// Stage 2 + loss: deaths_pred[i,m] = sum_{k=2..i} fmT[m][k]*pred[i-k,m];
// row 0 overridden to 1e-9; accumulate squared error, per-block partials.
// ---------------------------------------------------------------------------
__global__ __launch_bounds__(256) void stage2_kernel(
    const float* __restrict__ predT_f, const float* __restrict__ fmT,
    const float* __restrict__ gt, float* __restrict__ partial)
{
    __shared__ float red[256];
    int t = blockIdx.x * 256 + threadIdx.x;
    int m = t >> 12;           // column (uniform per 4096-thread span)
    int i = t & (N2 - 1);      // row (contiguous across lanes)
    const float* pm = predT_f + m * N2;
    const float* fm = fmT + m * N2;
    __builtin_prefetch(pm, 0, 1);  // global_prefetch_b8
    __builtin_prefetch(fm, 0, 1);
    float acc = 0.0f;
    #pragma unroll 4
    for (int k = 2; k <= i; ++k)
        acc = fmaf(fm[k], pm[i - k], acc);
    float dp = (i == 0) ? 1e-9f : acc;
    float e  = dp - gt[i * M + m];
    red[threadIdx.x] = e * e;
    __syncthreads();
    for (int s = 128; s > 0; s >>= 1) {
        if (threadIdx.x < s) red[threadIdx.x] += red[threadIdx.x + s];
        __syncthreads();
    }
    if (threadIdx.x == 0) partial[blockIdx.x] = red[0];
}

// Deterministic final reduction (fixed tree order -> graph-replay stable).
__global__ __launch_bounds__(1024) void final_kernel(
    const float* __restrict__ partial, float* __restrict__ out)
{
    __shared__ float red[1024];
    red[threadIdx.x] = partial[threadIdx.x];
    __syncthreads();
    for (int s = 512; s > 0; s >>= 1) {
        if (threadIdx.x < s) red[threadIdx.x] += red[threadIdx.x + s];
        __syncthreads();
    }
    if (threadIdx.x == 0)
        out[0] = red[0] * (1.0f / ((float)N2 * (float)M));
}

extern "C" void kernel_launch(void* const* d_in, const int* in_sizes, int n_in,
                              void* d_out, int out_size, void* d_ws, size_t ws_size,
                              hipStream_t stream) {
    const float* rt  = (const float*)d_in[0];   // (4096,64)
    const float* gt  = (const float*)d_in[1];   // (4096,64)
    const int*   idx = (const int*)d_in[2];     // (64,)
    const float* si  = (const float*)d_in[3];   // (4096,)
    const float* f   = (const float*)d_in[4];   // (4096,64)
    float* out = (float*)d_out;

    char* ws = (char*)d_ws;
    _Float16* predT_h = (_Float16*)(ws + 0);            // 64*4096*2 = 512 KB
    float*    predT_f = (float*)(ws + 524288);          // 1 MB
    _Float16* sirev   = (_Float16*)(ws + 1572864);      // 8 KB
    float*    fmT     = (float*)(ws + 1581056);         // 1 MB
    float*    partial = (float*)(ws + 2629632);         // 4 KB (1024 partials)

    stage1_kernel<<<1, 512, 0, stream>>>(rt, si, sirev, predT_f, predT_h);
    fmT_kernel<<<1024, 256, 0, stream>>>(f, idx, fmT);
    stage2_kernel<<<1024, 256, 0, stream>>>(predT_f, fmT, gt, partial);
    final_kernel<<<1, 1024, 0, stream>>>(partial, out);
}
